// LocationSemanticModel_79474074845271
// MI455X (gfx1250) — matrix operand order repaired and checked by reading, not compiled, not run on hardware
//
#include <hip/hip_runtime.h>
#include <hip/hip_bf16.h>

#define BB 256
#define LLEN 400
#define HH 256
#define NLOC 10000
#define ATT_H 10
#define G3 768      // 3*H
#define K2 512      // 2*H (concat [x|h])

typedef __attribute__((ext_vector_type(16))) __bf16 v16bf;
typedef __attribute__((ext_vector_type(8)))  float  v8f;
typedef __attribute__((ext_vector_type(4)))  unsigned int u32x4;
typedef __attribute__((ext_vector_type(8)))  int          i32x8;
typedef __attribute__((ext_vector_type(4)))  int          i32x4;

#if defined(__HIP_DEVICE_COMPILE__) && __has_builtin(__builtin_amdgcn_tensor_load_to_lds)
#define HAVE_TDM 1
#else
#define HAVE_TDM 0
#endif

union FragAB { v16bf v; unsigned int u[8]; u32x4 q[2]; };

__device__ __forceinline__ unsigned short f32_to_bf16(float f) {
  unsigned int u = __float_as_uint(f);
  unsigned int r = u + 0x7FFFu + ((u >> 16) & 1u);   // RNE
  return (unsigned short)(r >> 16);
}

// A fragment: 16x32 bf16 (MxK), ISA 7.12.2 layout.
// lane m = lane&15, k-half = lane>>4; per lane: two contiguous 16B chunks.
__device__ __forceinline__ v16bf load_frag_A(const unsigned short* p, int stride,
                                             int kbase, int lane) {
  FragAB f;
  int m  = lane & 15;
  int kh = (lane >> 4) << 3;
  const unsigned short* base = p + (long)m * stride + kbase + kh;
  f.q[0] = *(const u32x4*)(base);        // K = kbase+kh .. +7
  f.q[1] = *(const u32x4*)(base + 16);   // K = kbase+16+kh .. +7
  return f.v;
}

// B fragment: 32x16 bf16 (KxN), B[k][n] = W[rowbase+n][k]; per lane: two contiguous 16B chunks (same K layout as A).
__device__ __forceinline__ v16bf load_frag_B(const unsigned short* W, int stride,
                                             int rowbase, int kbase, int lane) {
  FragAB f;
  int n = lane & 15;
  const unsigned short* base =
      W + (long)(rowbase + n) * stride + kbase + ((lane >> 4) << 3);
  f.q[0] = *(const u32x4*)(base);
  f.q[1] = *(const u32x4*)(base + 16);
  return f.v;
}

__device__ __forceinline__ v8f wmma_bf16(v16bf a, v16bf b, v8f c) {
  return __builtin_amdgcn_wmma_f32_16x16x32_bf16(false, a, false, b, (short)0, c,
                                                 false, false);
}

__device__ __forceinline__ float sigmoidf_(float x) { return 1.f / (1.f + __expf(-x)); }

// ---------------- prep: bf16 weight conversion + sync counter reset -------------
__global__ void prep_kernel(const float* __restrict__ Wih, const float* __restrict__ Whh,
                            const float* __restrict__ fcW,
                            unsigned short* __restrict__ Wcat,
                            unsigned short* __restrict__ fcWb, int* __restrict__ sync) {
  long tid = (long)blockIdx.x * blockDim.x + threadIdx.x;
  long nth = (long)gridDim.x * blockDim.x;
  if (tid < 64) sync[tid] = 0;
  for (long i = tid; i < (long)G3 * K2; i += nth) {
    int row = (int)(i >> 9), k = (int)(i & 511);
    float v = (k < HH) ? Wih[row * HH + k] : Whh[row * HH + (k - HH)];
    Wcat[i] = f32_to_bf16(v);
  }
  for (long i = tid; i < (long)NLOC * HH; i += nth) fcWb[i] = f32_to_bf16(fcW[i]);
}

// ---------------- attention + rep + mask -> inp (bf16) --------------------------
__global__ void attn_kernel(const float* __restrict__ x, const int* __restrict__ locOH,
                            const int* __restrict__ valLen,
                            const float* __restrict__ loc_emb, const float* __restrict__ tim_emb,
                            const float* __restrict__ W1, const float* __restrict__ b1,
                            const float* __restrict__ W2, const float* __restrict__ b2,
                            unsigned short* __restrict__ inp_bf) {
  int wid  = (blockIdx.x << 2) + (threadIdx.x >> 5);  // one wave per (b,l)
  int lane = threadIdx.x & 31;
  int b = wid / LLEN, l = wid - b * LLEN;
  int loc  = locOH[b * LLEN + l];
  int slot = (int)x[((long)(b * LLEN + l)) * 5 + 4];
  const float* lrow = loc_emb + (long)loc * HH;
  const float* trow = tim_emb + (long)slot * HH;
  float lv[8], tv[8];
#pragma unroll
  for (int i = 0; i < 8; ++i) { int h = lane + (i << 5); lv[i] = lrow[h]; tv[i] = trow[h]; }
  float sL = 0.f, sT = 0.f;
  for (int c = 0; c < ATT_H; ++c) {
    float dl = 0.f, dt = 0.f;
#pragma unroll
    for (int i = 0; i < 8; ++i) {
      int h = lane + (i << 5);
      float w = W1[h * ATT_H + c];
      dl += lv[i] * w; dt += tv[i] * w;
    }
#pragma unroll
    for (int off = 16; off > 0; off >>= 1) {   // wave32 reduce
      dl += __shfl_xor(dl, off, 32);
      dt += __shfl_xor(dt, off, 32);
    }
    float w2 = W2[c];
    sL += fmaxf(dl + b1[c], 0.f) * w2;
    sT += fmaxf(dt + b1[c], 0.f) * w2;
  }
  sL += b2[0]; sT += b2[0];
  float mx = fmaxf(sL, sT);
  float e0 = __expf(sL - mx), e1 = __expf(sT - mx);
  float inv = 1.f / (e0 + e1);
  float w0 = e0 * inv, w1 = e1 * inv;
  float maskf = (l < valLen[b]) ? 1.f : 0.f;
  unsigned short* orow = inp_bf + ((long)(b * LLEN + l)) * HH;
#pragma unroll
  for (int i = 0; i < 8; ++i) {
    int h = lane + (i << 5);
    orow[h] = f32_to_bf16((w0 * lv[i] + w1 * tv[i]) * maskf);
  }
}

// ---------------- GRU: VGPR-resident weights, TDM x-staging, WMMA ---------------
// 64 WGs = 16 batch-groups (16 rows) x 4 H-slices (64 cols). Weights preloaded
// once into registers (24 v16bf B-fragments per wave); 400-step recurrence does
// only A-fragment LDS reads + 24 straight-line WMMAs per step.
__global__ void __launch_bounds__(256, 1)
gru_kernel(const unsigned short* __restrict__ inp_bf,
           const unsigned short* __restrict__ Wcat,
           const float* __restrict__ bih, const float* __restrict__ bhh,
           float* __restrict__ hidden, int* __restrict__ sync) {
  __shared__ unsigned short Ast[16 * 512];   // [16][x(256)|h(256)] bf16, 1024B row
  __shared__ float gates[16 * 256];          // r|z|inn|hn x 64 cols
  __shared__ float bihs[G3], bhhs[G3];

  int tid = threadIdx.x, lane = tid & 31, w = tid >> 5;
  int grp = blockIdx.x >> 2, slice = blockIdx.x & 3;
  int b0 = grp << 4;

  for (int i = tid; i < G3; i += 256) { bihs[i] = bih[i]; bhhs[i] = bhh[i]; }

  // wave tile assignment: heavy = r/z (K=512); light = inn (K<256) or hn (K>=256)
  int hg = w >> 2, hst = w & 3;
  int heavyGRow = hg * HH + slice * 64 + hst * 16;   // gate-r/z rows of Wcat
  int lightGRow = 2 * HH + slice * 64 + hst * 16;    // gate-n rows
  int lightIsHN = (w >> 2) & 1;
  int lkb = lightIsHN << 8;                          // light K base: 0 or 256

  // Preload all B fragments into VGPRs (global, reused 400x)
  v16bf bH[16], bL[8];
#pragma unroll
  for (int kk = 0; kk < 16; ++kk) bH[kk] = load_frag_B(Wcat, K2, heavyGRow, kk << 5, lane);
#pragma unroll
  for (int kk = 0; kk < 8; ++kk)  bL[kk] = load_frag_B(Wcat, K2, lightGRow, lkb + (kk << 5), lane);
  __syncthreads();

  for (int t = 0; t < LLEN; ++t) {
    // ---- stage A = [x_t | h_{t-1}] ----
#if HAVE_TDM
    if (w == 0) {   // one TDM per WG: 16 rows x 512B, +512B LDS pad -> row stride 1024B
      u32x4 g0;
      unsigned long ga = (unsigned long)(const void*)(inp_bf + ((long)b0 * LLEN + t) * HH);
      g0[0] = 1u;                                           // count=1
      g0[1] = (unsigned int)(size_t)(void*)Ast;             // lds_addr (bytes)
      g0[2] = (unsigned int)ga;                             // global_addr[31:0]
      g0[3] = (unsigned int)((ga >> 32) & 0x01FFFFFFu) | (2u << 30); // addr[56:32]|type=2
      i32x8 g1;
      g1[0] = (int)((1u << 16) | (1u << 20) | (6u << 22) | (127u << 25)); // 2B, pad 128DW/128DW
      g1[1] = (int)((256u & 0xFFFFu) << 16);                // tensor_dim0 lo
      g1[2] = (int)((16u & 0xFFFFu) << 16);                 // dim0 hi=0 | tensor_dim1 lo=16
      g1[3] = (int)(256u << 16);                            // dim1 hi=0 | tile_dim0=256
      g1[4] = (int)(16u);                                   // tile_dim1=16 | tile_dim2=0
      g1[5] = (int)(LLEN * HH);                             // tensor_dim0_stride = 102400
      g1[6] = 0; g1[7] = 0;
      i32x4 gz = {0, 0, 0, 0};
#if defined(__clang_major__) && __clang_major__ >= 23
      i32x8 gz8 = {0, 0, 0, 0, 0, 0, 0, 0};
      __builtin_amdgcn_tensor_load_to_lds(g0, g1, gz, gz, gz8, 0);
#else
      __builtin_amdgcn_tensor_load_to_lds(g0, g1, gz, gz, 0);
#endif
    }
#else
    for (int i = tid; i < 2048; i += 256) {                 // manual x copy (uints)
      int m = i >> 7, kp = i & 127;
      ((unsigned int*)Ast)[(m << 8) + kp] =
          ((const unsigned int*)inp_bf)[((long)(b0 + m) * LLEN + t) * 128 + kp];
    }
#endif
    for (int i = tid; i < 4096; i += 256) {                 // h part, f32 -> bf16
      int m = i >> 8, k = i & 255;
      float hv = (t == 0) ? 0.f : hidden[((long)(b0 + m) * LLEN + (t - 1)) * HH + k];
      Ast[(m << 9) + 256 + k] = f32_to_bf16(hv);
    }
#if HAVE_TDM
    if (w == 0) __builtin_amdgcn_s_wait_tensorcnt(0);
#endif
    __syncthreads();

    // ---- 24 straight-line WMMAs, weights from VGPRs ----
    v8f accH = {}, accL = {};
#pragma unroll
    for (int kk = 0; kk < 16; ++kk)
      accH = wmma_bf16(load_frag_A(Ast, 512, kk << 5, lane), bH[kk], accH);
#pragma unroll
    for (int kk = 0; kk < 8; ++kk)
      accL = wmma_bf16(load_frag_A(Ast, 512, lkb + (kk << 5), lane), bL[kk], accL);

    // spill tiles: cols 0-63 r | 64-127 z | 128-191 inn | 192-255 hn
    {
      int n = lane & 15, mh = (lane >> 4) << 3;
      int colH = hg * 64 + hst * 16 + n;
      int colL = 128 + lightIsHN * 64 + hst * 16 + n;
#pragma unroll
      for (int r = 0; r < 8; ++r) {
        gates[((mh + r) << 8) + colH] = accH[r];
        gates[((mh + r) << 8) + colL] = accL[r];
      }
    }
    __syncthreads();

    // ---- gate math for 16 rows x 64 h-cols ----
    for (int e = tid; e < 1024; e += 256) {
      int m = e >> 6, j = e & 63;
      int hc = slice * 64 + j;
      float rg = sigmoidf_(gates[(m << 8) + j]      + bihs[hc]      + bhhs[hc]);
      float zg = sigmoidf_(gates[(m << 8) + 64 + j] + bihs[HH + hc] + bhhs[HH + hc]);
      float nn = tanhf(gates[(m << 8) + 128 + j] + bihs[2 * HH + hc] +
                       rg * (gates[(m << 8) + 192 + j] + bhhs[2 * HH + hc]));
      float hp = (t == 0) ? 0.f : hidden[((long)(b0 + m) * LLEN + (t - 1)) * HH + hc];
      hidden[((long)(b0 + m) * LLEN + t) * HH + hc] = (1.f - zg) * nn + zg * hp;
    }
    __syncthreads();

    // ---- barrier among the 4 slice-WGs of this batch group ----
    if (tid == 0) {
      __hip_atomic_fetch_add(&sync[grp], 1, __ATOMIC_RELEASE, __HIP_MEMORY_SCOPE_AGENT);
      while (__hip_atomic_load(&sync[grp], __ATOMIC_ACQUIRE, __HIP_MEMORY_SCOPE_AGENT) <
             4 * (t + 1))
        __builtin_amdgcn_s_sleep(2);
    }
    __syncthreads();
  }
}

// ---------------- spatio-temporal weights + weighted hidden reduction -----------
__global__ void stw_kernel(const float* __restrict__ x, const int* __restrict__ valLen,
                           const float* __restrict__ hidden,
                           float* __restrict__ sum_f32, unsigned short* __restrict__ sum_bf) {
  __shared__ float stw[LLEN];
  __shared__ float red[256];
  int b = blockIdx.x, tid = threadIdx.x;
  int last = valLen[b] - 1;
  const float* xb = x + (long)b * LLEN * 5;
  float last_t = xb[last * 5 + 2];
  float ls0 = xb[last * 5 + 0], ls1 = xb[last * 5 + 1];
  float nlast = sqrtf(ls0 * ls0 + ls1 * ls1 + 1e-12f);
  int vl = last + 1;
  for (int l = tid; l < LLEN; l += 256) {
    float mf = (l < vl) ? 1.f : 0.f;
    float minus = (last_t - xb[l * 5 + 2] * mf) * (1.f / 86400.f);
    float wgh = (minus != last_t) ? __expf(-minus * 0.1f) : 0.f;
    float d0 = ls0 - xb[l * 5 + 0] * mf, d1 = ls1 - xb[l * 5 + 1] * mf;
    float d = sqrtf(d0 * d0 + d1 * d1 + 1e-12f);
    float wgs = (d != nlast) ? __expf(-d * 0.01f) : 0.f;
    stw[l] = wgh * wgs;
  }
  __syncthreads();
  float part = 0.f;
  for (int l = tid; l < LLEN; l += 256) part += stw[l];
  red[tid] = part; __syncthreads();
  for (int s = 128; s > 0; s >>= 1) { if (tid < s) red[tid] += red[tid + s]; __syncthreads(); }
  float inv = 1.f / red[0];
  float acc = 0.f;
  const float* hb = hidden + (long)b * LLEN * HH;
  for (int l = 0; l < LLEN; ++l) acc += stw[l] * hb[(long)l * HH + tid];
  acc *= inv;
  sum_f32[b * HH + tid] = acc;
  sum_bf[b * HH + tid] = f32_to_bf16(acc);
}

// ---------------- final FC: (256 x 10000 x 256) WMMA GEMM + gathered bias -------
__global__ void fc_kernel(const unsigned short* __restrict__ sum_bf,
                          const unsigned short* __restrict__ fcWb,
                          const float* __restrict__ fc_b, const float* __restrict__ spa_e,
                          const int* __restrict__ locOH, const int* __restrict__ valLen,
                          float* __restrict__ out) {
  __shared__ unsigned short Ast[16 * 256];
  int tid = threadIdx.x, lane = tid & 31, w = tid >> 5;
  int m0 = blockIdx.y << 4;
  int n0 = (blockIdx.x << 7) + (w << 4);
  for (int i = tid; i < 16 * 128; i += 256)
    ((unsigned int*)Ast)[i] = ((const unsigned int*)sum_bf)[(m0 << 7) + i];
  __syncthreads();
  if (n0 < NLOC) {                                          // wave-uniform guard
    v8f acc = {};
    int n = lane & 15;
#pragma unroll
    for (int kk = 0; kk < 8; ++kk) {
      int kbase = kk << 5;
      if (kk < 7)
        __builtin_prefetch(fcWb + (long)(n0 + n) * HH + kbase + 32, 0, 1);
      v16bf a = load_frag_A(Ast, 256, kbase, lane);
      v16bf b = load_frag_B(fcWb, HH, n0, kbase, lane);
      acc = wmma_bf16(a, b, acc);
    }
    int mh = (lane >> 4) << 3;
#pragma unroll
    for (int r = 0; r < 8; ++r) {
      int b2 = m0 + mh + r, nn = n0 + n;
      int lastp = locOH[b2 * LLEN + (valLen[b2] - 1)];
      out[(long)b2 * NLOC + nn] = acc[r] + fc_b[nn] + spa_e[(long)lastp * NLOC + nn];
    }
  }
}

extern "C" void kernel_launch(void* const* d_in, const int* in_sizes, int n_in,
                              void* d_out, int out_size, void* d_ws, size_t ws_size,
                              hipStream_t stream) {
  (void)in_sizes; (void)n_in; (void)out_size; (void)ws_size;
  const float* x      = (const float*)d_in[0];
  const int*   locOH  = (const int*)d_in[2];
  const int*   valLen = (const int*)d_in[3];
  const float* spa_e  = (const float*)d_in[4];
  const float* loc_e  = (const float*)d_in[5];
  const float* tim_e  = (const float*)d_in[6];
  const float* W1     = (const float*)d_in[7];
  const float* b1     = (const float*)d_in[8];
  const float* W2     = (const float*)d_in[9];
  const float* b2     = (const float*)d_in[10];
  const float* Wih    = (const float*)d_in[11];
  const float* Whh    = (const float*)d_in[12];
  const float* bih    = (const float*)d_in[13];
  const float* bhh    = (const float*)d_in[14];
  const float* fcW    = (const float*)d_in[15];
  const float* fcb    = (const float*)d_in[16];
  float* out = (float*)d_out;

  unsigned char* ws = (unsigned char*)d_ws;
  size_t off = 0;
  auto alloc = [&](size_t bytes) -> void* {
    void* p = ws + off;
    off = (off + bytes + 255) & ~(size_t)255;
    return p;
  };
  unsigned short* inp_bf  = (unsigned short*)alloc((size_t)BB * LLEN * HH * 2); // 52 MB
  float*          hidden  = (float*)alloc((size_t)BB * LLEN * HH * 4);          // 105 MB
  unsigned short* Wcat    = (unsigned short*)alloc((size_t)G3 * K2 * 2);
  unsigned short* fcWb    = (unsigned short*)alloc((size_t)NLOC * HH * 2);
  float*          sum_f32 = (float*)alloc((size_t)BB * HH * 4);
  unsigned short* sum_bf  = (unsigned short*)alloc((size_t)BB * HH * 2);
  int*            syncp   = (int*)alloc(64 * sizeof(int));

  prep_kernel<<<1024, 256, 0, stream>>>(Wih, Whh, fcW, Wcat, fcWb, syncp);
  attn_kernel<<<(BB * LLEN) / 4, 128, 0, stream>>>(x, locOH, valLen, loc_e, tim_e,
                                                   W1, b1, W2, b2, inp_bf);
  gru_kernel<<<64, 256, 0, stream>>>(inp_bf, Wcat, bih, bhh, hidden, syncp);
  stw_kernel<<<BB, 256, 0, stream>>>(x, valLen, hidden, sum_f32, sum_bf);
  fc_kernel<<<dim3((NLOC + 127) / 128, BB / 16), 256, 0, stream>>>(
      sum_bf, fcWb, fcb, spa_e, locOH, valLen, out);
}
